// SimpleLLMDecodeModel_35442070127136
// MI455X (gfx1250) — compile-verified, matrix operand order checked by reading, not run on hardware
//
#include <hip/hip_runtime.h>
#include <hip/hip_bf16.h>

typedef __attribute__((ext_vector_type(16))) __bf16 v16bf;
typedef __attribute__((ext_vector_type(8)))  __bf16 v8bf;
typedef __attribute__((ext_vector_type(8)))  float  v8f;
typedef __attribute__((ext_vector_type(4)))  int    v4i;

#define AS1 __attribute__((address_space(1)))
#define AS3 __attribute__((address_space(3)))

#if __has_builtin(__builtin_amdgcn_global_load_async_to_lds_b128) && \
    __has_builtin(__builtin_amdgcn_s_wait_asynccnt)
#define USE_ASYNC_LDS 1
#else
#define USE_ASYNC_LDS 0
#endif

__device__ __forceinline__ __bf16 f2bf(float f) {
    union { float f; unsigned u; } v; v.f = f;
    unsigned r = v.u + 0x7FFFu + ((v.u >> 16) & 1u);   // round-to-nearest-even
    unsigned short h = (unsigned short)(r >> 16);
    return __builtin_bit_cast(__bf16, h);
}

// ---------------------------------------------------------------------------
// Prep kernels (bandwidth-bound, trivial)
// ---------------------------------------------------------------------------
__global__ void cvt_f32_to_bf16(const float* __restrict__ in,
                                __bf16* __restrict__ out, size_t n) {
    size_t i = (size_t)blockIdx.x * blockDim.x + threadIdx.x;
    size_t stride = (size_t)gridDim.x * blockDim.x;
    for (; i < n; i += stride) out[i] = f2bf(in[i]);
}

// kb: [112][4096] bf16; rows 0..99 = k_cache, rows 100..111 = 0
__global__ void prep_k(const float* __restrict__ kc, __bf16* __restrict__ kb) {
    size_t i = (size_t)blockIdx.x * blockDim.x + threadIdx.x;
    if (i >= (size_t)112 * 4096) return;
    int r = (int)(i >> 12);
    int c = (int)(i & 4095);
    kb[i] = (r < 100) ? f2bf(kc[(size_t)r * 4096 + c])
                      : __builtin_bit_cast(__bf16, (unsigned short)0);
}

// vtb: [4096][128] bf16; vtb[n][l] = v_cache[l][n] for l<100 else 0
__global__ void prep_v(const float* __restrict__ vc, __bf16* __restrict__ vtb) {
    size_t i = (size_t)blockIdx.x * blockDim.x + threadIdx.x;
    if (i >= (size_t)4096 * 128) return;
    int n = (int)(i >> 7);
    int l = (int)(i & 127);
    vtb[i] = (l < 100) ? f2bf(vc[(size_t)l * 4096 + n])
                       : __builtin_bit_cast(__bf16, (unsigned short)0);
}

// ---------------------------------------------------------------------------
// Tiled WMMA bf16 GEMM:  C[M,N] = scale * (A[M,K] @ Brows[N,K]^T) + bias[N]
// Block: 256 threads (8 waves). Tile: 128x128, BK=32. Wave tile: 64x32.
// GUARD=false: async double-buffered LDS copies (no bounds checks; dims must
//              be tile multiples). GUARD=true: sync path with N-bounds checks.
// blockIdx.z K-split: each z-slab writes to outF + z*M*ldOut (partials).
// ---------------------------------------------------------------------------
#define BM 128
#define BN 128
#define BK 32
#define LDSS 40   // padded LDS row stride in elements (80 bytes)

__device__ __forceinline__ void compute_tile(const __bf16* AsP, const __bf16* BsP,
                                             v8f (&acc)[4][2], int waveM, int waveN,
                                             int fr, int fk) {
    v16bf bfrag[2];
    #pragma unroll
    for (int j = 0; j < 2; ++j) {
        int brow = waveN * 32 + j * 16 + fr;
        v8bf lo = *(const v8bf*)(BsP + brow * LDSS + fk);
        v8bf hi = *(const v8bf*)(BsP + brow * LDSS + fk + 16);
        #pragma unroll
        for (int e = 0; e < 8; ++e) { bfrag[j][e] = lo[e]; bfrag[j][e + 8] = hi[e]; }
    }
    #pragma unroll
    for (int i = 0; i < 4; ++i) {
        int arow = waveM * 64 + i * 16 + fr;
        v8bf lo = *(const v8bf*)(AsP + arow * LDSS + fk);
        v8bf hi = *(const v8bf*)(AsP + arow * LDSS + fk + 16);
        v16bf afrag;
        #pragma unroll
        for (int e = 0; e < 8; ++e) { afrag[e] = lo[e]; afrag[e + 8] = hi[e]; }
        #pragma unroll
        for (int j = 0; j < 2; ++j) {
            acc[i][j] = __builtin_amdgcn_wmma_f32_16x16x32_bf16(
                false, afrag, false, bfrag[j], (short)0, acc[i][j], false, false);
        }
    }
}

template <bool GUARD>
__global__ __launch_bounds__(256)
void gemm_bf16_wmma(const __bf16* __restrict__ A,
                    const __bf16* __restrict__ Brows,
                    const float*  __restrict__ bias,
                    float scale,
                    float*  __restrict__ outF,
                    __bf16* __restrict__ outB,
                    int M, int N, int K, int ldOut, int kChunk)
{
    __shared__ __bf16 As[2][BM * LDSS];
    __shared__ __bf16 Bs[2][BN * LDSS];

    const int tid   = threadIdx.x;
    const int lane  = tid & 31;
    const int wave  = tid >> 5;
    const int waveM = wave & 1;          // 2 waves along M
    const int waveN = wave >> 1;         // 4 waves along N
    const int blockM = blockIdx.y * BM;
    const int blockN = blockIdx.x * BN;

    const int kStart = blockIdx.z * kChunk;
    const int kEnd0  = kStart + kChunk;
    const int kEnd   = kEnd0 < K ? kEnd0 : K;

    v8f acc[4][2];
    #pragma unroll
    for (int i = 0; i < 4; ++i)
        #pragma unroll
        for (int j = 0; j < 2; ++j)
            acc[i][j] = (v8f){0.f, 0.f, 0.f, 0.f, 0.f, 0.f, 0.f, 0.f};

    // fragment lane addressing (16-bit operand, 16x32 layout)
    const int fr = lane & 15;            // row within 16-row tile
    const int fk = (lane >> 4) * 8;      // K sub-offset: 0 or 8

    // cooperative-copy coordinates: 512 x 16B chunks per tile, 2 per thread
    const int row0 = tid >> 2;           // 0..63
    const int row1 = row0 + 64;          // 64..127
    const int kc   = (tid & 3) * 8;      // element offset within BK

    if (!GUARD) {
#if USE_ASYNC_LDS
        // --------- async double-buffered pipeline ---------
#define ASYNC_COPY_TILE(buf, k0)                                               \
        do {                                                                   \
            __builtin_amdgcn_global_load_async_to_lds_b128(                    \
                (AS1 v4i*)(A + (size_t)(blockM + row0) * K + (k0) + kc),       \
                (AS3 v4i*)(&As[buf][row0 * LDSS + kc]), 0, 0);                 \
            __builtin_amdgcn_global_load_async_to_lds_b128(                    \
                (AS1 v4i*)(A + (size_t)(blockM + row1) * K + (k0) + kc),       \
                (AS3 v4i*)(&As[buf][row1 * LDSS + kc]), 0, 0);                 \
            __builtin_amdgcn_global_load_async_to_lds_b128(                    \
                (AS1 v4i*)(Brows + (size_t)(blockN + row0) * K + (k0) + kc),   \
                (AS3 v4i*)(&Bs[buf][row0 * LDSS + kc]), 0, 0);                 \
            __builtin_amdgcn_global_load_async_to_lds_b128(                    \
                (AS1 v4i*)(Brows + (size_t)(blockN + row1) * K + (k0) + kc),   \
                (AS3 v4i*)(&Bs[buf][row1 * LDSS + kc]), 0, 0);                 \
        } while (0)

        ASYNC_COPY_TILE(0, kStart);
        const int nIter = (kEnd - kStart) / BK;
        for (int it = 0; it < nIter; ++it) {
            const int k0  = kStart + it * BK;
            const int cur = it & 1;
            const int nxt = cur ^ 1;
            if (it + 1 < nIter) {
                ASYNC_COPY_TILE(nxt, k0 + BK);
                __builtin_amdgcn_s_wait_asynccnt(4);  // prev tile's 4 done (in-order)
            } else {
                __builtin_amdgcn_s_wait_asynccnt(0);
            }
            __syncthreads();
            compute_tile(&As[cur][0], &Bs[cur][0], acc, waveM, waveN, fr, fk);
            __syncthreads();                          // protect cur before reuse
        }
#undef ASYNC_COPY_TILE
#else
        // --------- sync fallback (no guards) ---------
        for (int k0 = kStart; k0 < kEnd; k0 += BK) {
            v8bf a0 = *(const v8bf*)(A + (size_t)(blockM + row0) * K + k0 + kc);
            v8bf a1 = *(const v8bf*)(A + (size_t)(blockM + row1) * K + k0 + kc);
            v8bf b0 = *(const v8bf*)(Brows + (size_t)(blockN + row0) * K + k0 + kc);
            v8bf b1 = *(const v8bf*)(Brows + (size_t)(blockN + row1) * K + k0 + kc);
            *(v8bf*)(&As[0][row0 * LDSS + kc]) = a0;
            *(v8bf*)(&As[0][row1 * LDSS + kc]) = a1;
            *(v8bf*)(&Bs[0][row0 * LDSS + kc]) = b0;
            *(v8bf*)(&Bs[0][row1 * LDSS + kc]) = b1;
            __syncthreads();
            compute_tile(&As[0][0], &Bs[0][0], acc, waveM, waveN, fr, fk);
            __syncthreads();
        }
#endif
    } else {
        // --------- guarded sync path (small GEMMs only) ---------
        for (int k0 = kStart; k0 < kEnd; k0 += BK) {
            v8bf a0 = {}, a1 = {}, b0 = {}, b1 = {};
            if (blockM + row0 < M) a0 = *(const v8bf*)(A + (size_t)(blockM + row0) * K + k0 + kc);
            if (blockM + row1 < M) a1 = *(const v8bf*)(A + (size_t)(blockM + row1) * K + k0 + kc);
            if (blockN + row0 < N) b0 = *(const v8bf*)(Brows + (size_t)(blockN + row0) * K + k0 + kc);
            if (blockN + row1 < N) b1 = *(const v8bf*)(Brows + (size_t)(blockN + row1) * K + k0 + kc);
            *(v8bf*)(&As[0][row0 * LDSS + kc]) = a0;
            *(v8bf*)(&As[0][row1 * LDSS + kc]) = a1;
            *(v8bf*)(&Bs[0][row0 * LDSS + kc]) = b0;
            *(v8bf*)(&Bs[0][row1 * LDSS + kc]) = b1;
            __syncthreads();
            compute_tile(&As[0][0], &Bs[0][0], acc, waveM, waveN, fr, fk);
            __syncthreads();
        }
    }

    // K-split partial slab
    float*  outFz = outF ? outF + (size_t)blockIdx.z * (size_t)M * (size_t)ldOut : nullptr;
    __bf16* outBz = outB ? outB + (size_t)blockIdx.z * (size_t)M * (size_t)ldOut : nullptr;

    // C layout: VGPR r -> row = r + (lane>=16 ? 8 : 0); col = lane & 15
    const int rsel = (lane >> 4) * 8;
    const int ccol = lane & 15;
    #pragma unroll
    for (int i = 0; i < 4; ++i) {
        #pragma unroll
        for (int j = 0; j < 2; ++j) {
            int col = blockN + waveN * 32 + j * 16 + ccol;
            if (GUARD && col >= N) continue;
            float b = bias ? bias[col] : 0.f;
            #pragma unroll
            for (int r = 0; r < 8; ++r) {
                int row = blockM + waveM * 64 + i * 16 + rsel + r;
                if (GUARD && row >= M) continue;
                float val = acc[i][j][r] * scale + b;
                if (outFz) outFz[(size_t)row * ldOut + col] = val;
                else       outBz[(size_t)row * ldOut + col] = f2bf(val);
            }
        }
    }
}

// ---------------------------------------------------------------------------
// Softmax: sum 4 K-split partial slabs of scores [4][8192][112], softmax over
// 101 valid columns -> weights bf16 [8192][128] (cols >= 101 zeroed)
// ---------------------------------------------------------------------------
__global__ __launch_bounds__(128)
void softmax_kernel(const float* __restrict__ sc, __bf16* __restrict__ wb) {
    const int row = blockIdx.x;
    const int tid = threadIdx.x;
    __shared__ float redM[4];
    __shared__ float redS[4];
    const size_t slab = (size_t)8192 * 112;

    float v = -1e30f;
    if (tid < 101) {
        size_t idx = (size_t)row * 112 + tid;
        v = sc[idx] + sc[idx + slab] + sc[idx + 2 * slab] + sc[idx + 3 * slab];
    }
    float m = v;
    #pragma unroll
    for (int off = 16; off; off >>= 1) m = fmaxf(m, __shfl_xor(m, off, 32));
    if ((tid & 31) == 0) redM[tid >> 5] = m;
    __syncthreads();
    float rowmax = fmaxf(fmaxf(redM[0], redM[1]), fmaxf(redM[2], redM[3]));

    float e = (tid < 101) ? __expf(v - rowmax) : 0.f;
    float s = e;
    #pragma unroll
    for (int off = 16; off; off >>= 1) s += __shfl_xor(s, off, 32);
    if ((tid & 31) == 0) redS[tid >> 5] = s;
    __syncthreads();
    float tot = redS[0] + redS[1] + redS[2] + redS[3];

    wb[(size_t)row * 128 + tid] = f2bf(e / tot);
}

// ---------------------------------------------------------------------------
// Launch
// ---------------------------------------------------------------------------
extern "C" void kernel_launch(void* const* d_in, const int* in_sizes, int n_in,
                              void* d_out, int out_size, void* d_ws, size_t ws_size,
                              hipStream_t stream) {
    const float* x       = (const float*)d_in[0];
    const float* q_w     = (const float*)d_in[1];
    const float* q_b     = (const float*)d_in[2];
    const float* k_cache = (const float*)d_in[3];
    const float* v_cache = (const float*)d_in[4];
    const float* out_w   = (const float*)d_in[5];
    const float* out_b   = (const float*)d_in[6];
    float* out = (float*)d_out;

    char* ws = (char*)d_ws;
    const size_t MB = 1ull << 20;
    __bf16* xb    = (__bf16*)(ws + 0);           // 64 MB  [8192,4096]
    __bf16* attnb = xb;                          // alias: xb dead after GEMM1
    __bf16* qb    = (__bf16*)(ws + 64 * MB);     // 64 MB  [8192,4096]
    __bf16* wslot = (__bf16*)(ws + 128 * MB);    // 32 MB  q_w then out_w bf16
    __bf16* kb    = (__bf16*)(ws + 160 * MB);    // 1 MB   [112,4096]
    __bf16* vtb   = (__bf16*)(ws + 161 * MB);    // 1 MB   [4096,128]
    float*  sc    = (float*)(ws + 162 * MB);     // 16 MB  [4][8192,112]
    __bf16* wb    = (__bf16*)(ws + 178 * MB);    // 2 MB   [8192,128]

    const size_t nX = (size_t)8192 * 4096;
    const size_t nW = (size_t)4096 * 4096;

    cvt_f32_to_bf16<<<4096, 256, 0, stream>>>(x, xb, nX);
    cvt_f32_to_bf16<<<4096, 256, 0, stream>>>(q_w, wslot, nW);
    prep_k<<<(112 * 4096 + 255) / 256, 256, 0, stream>>>(k_cache, kb);
    prep_v<<<(4096 * 128 + 255) / 256, 256, 0, stream>>>(v_cache, vtb);

    dim3 blk(256);
    // 1) q = x @ q_w^T + q_b  -> bf16
    gemm_bf16_wmma<false><<<dim3(4096 / BN, 8192 / BM, 1), blk, 0, stream>>>(
        xb, wslot, q_b, 1.0f, nullptr, qb, 8192, 4096, 4096, 4096, 4096);
    // 2) scores = (q @ k^T) * 1/sqrt(128) -> 4 partial f32 slabs [8192,112]
    gemm_bf16_wmma<true><<<dim3(1, 8192 / BM, 4), blk, 0, stream>>>(
        qb, kb, nullptr, 0.08838834764831845f, sc, nullptr, 8192, 112, 4096, 112, 1024);
    // 3) softmax (sums slabs) -> bf16 weights [8192,128]
    softmax_kernel<<<8192, 128, 0, stream>>>(sc, wb);
    // convert out_w now that GEMM1 has consumed wslot
    cvt_f32_to_bf16<<<4096, 256, 0, stream>>>(out_w, wslot, nW);
    // 4) attn = weights @ v  (v transposed into [4096,128]) -> bf16
    gemm_bf16_wmma<false><<<dim3(4096 / BN, 8192 / BM, 1), blk, 0, stream>>>(
        wb, vtb, nullptr, 1.0f, nullptr, attnb, 8192, 4096, 128, 4096, 128);
    // 5) out = attn @ out_w^T + out_b -> f32
    gemm_bf16_wmma<false><<<dim3(4096 / BN, 8192 / BM, 1), blk, 0, stream>>>(
        attnb, wslot, out_b, 1.0f, out, nullptr, 8192, 4096, 4096, 4096, 4096);
}
